// TemporalCrossAttention_33045478375699
// MI455X (gfx1250) — matrix operand order repaired and checked by reading, not compiled, not run on hardware
//
#include <hip/hip_runtime.h>

#define DD 512
#define TT 512
#define BB 32
#define HH 8
#define BT 16384  // BB*TT

typedef __attribute__((ext_vector_type(16))) __bf16 v16bf;
typedef __attribute__((ext_vector_type(8)))  float  v8f;

union Frag { uint4 u[2]; v16bf v; };

__device__ __forceinline__ unsigned short f2bf(float x) {
    unsigned int u = __float_as_uint(x);
    u += 0x7FFFu + ((u >> 16) & 1u);          // round-to-nearest-even
    return (unsigned short)(u >> 16);
}

__device__ __forceinline__ v8f wmma_bf16(const Frag& a, const Frag& b, v8f c) {
    return __builtin_amdgcn_wmma_f32_16x16x32_bf16(
        false, a.v, false, b.v, (short)0, c, false, false);
}

// ---------------------------------------------------------------------------
// Stage 1: QKV projections.  C[16384x512] = X @ W^T (+bias), scattered into
// q/k/v bf16 workspace with layout [b][h][t][128] (face half at d 0..63,
// skeleton half at d 64..127).  grid = (1024, 6), block = 256.
// ---------------------------------------------------------------------------
__global__ __launch_bounds__(256)
void qkv_proj_kernel(const float* __restrict__ xf, const float* __restrict__ xs,
                     const float* __restrict__ wq1, const float* __restrict__ wk1,
                     const float* __restrict__ wv1, const float* __restrict__ wq2,
                     const float* __restrict__ wk2, const float* __restrict__ wv2,
                     const float* __restrict__ bq1, const float* __restrict__ bk1,
                     const float* __restrict__ bv1, const float* __restrict__ bq2,
                     const float* __restrict__ bk2, const float* __restrict__ bv2,
                     unsigned short* __restrict__ qd, unsigned short* __restrict__ kd,
                     unsigned short* __restrict__ vd)
{
    __shared__ __align__(16) unsigned short As[128][40];
    __shared__ __align__(16) unsigned short Bs[64][40];

    const int g = blockIdx.y;
    const float* X = (g < 3) ? xf : xs;
    const float* W; const float* bias; unsigned short* dst;
    switch (g) {
        case 0:  W = wq1; bias = bq1; dst = qd; break;
        case 1:  W = wk1; bias = bk1; dst = kd; break;
        case 2:  W = wv1; bias = bv1; dst = vd; break;
        case 3:  W = wq2; bias = bq2; dst = qd; break;
        case 4:  W = wk2; bias = bk2; dst = kd; break;
        default: W = wv2; bias = bv2; dst = vd; break;
    }
    const int hoff = (g < 3) ? 0 : 64;

    const int M0 = (blockIdx.x >> 3) * 128;
    const int N0 = (blockIdx.x & 7) * 64;

    const int tid  = threadIdx.x;
    const int lane = tid & 31;
    const int wid  = tid >> 5;
    const int wm   = wid >> 1;   // 0..3 -> M offset 32*wm
    const int wn   = wid & 1;    // 0..1 -> N offset 32*wn

    const v8f vz = {0.f,0.f,0.f,0.f,0.f,0.f,0.f,0.f};
    v8f acc[2][2] = {{vz, vz}, {vz, vz}};

    const int r   = lane & 15;
    const int kb  = (lane >> 4) * 8;
    const int kb2 = (lane >> 4) * 16;

    for (int k0 = 0; k0 < DD; k0 += 32) {
        __syncthreads();
#pragma unroll
        for (int i = 0; i < 4; ++i) {             // A: 128x32 fp32 -> bf16
            int idx = tid + i * 256;
            int row = idx >> 3, c4 = idx & 7;
            float4 f = *(const float4*)(X + (size_t)(M0 + row) * DD + k0 + c4 * 4);
            unsigned int p0 = (unsigned int)f2bf(f.x) | ((unsigned int)f2bf(f.y) << 16);
            unsigned int p1 = (unsigned int)f2bf(f.z) | ((unsigned int)f2bf(f.w) << 16);
            *(uint2*)&As[row][c4 * 4] = make_uint2(p0, p1);
        }
#pragma unroll
        for (int i = 0; i < 2; ++i) {             // B: 64x32 fp32 -> bf16 (rows of W)
            int idx = tid + i * 256;
            int row = idx >> 3, c4 = idx & 7;
            float4 f = *(const float4*)(W + (size_t)(N0 + row) * DD + k0 + c4 * 4);
            unsigned int p0 = (unsigned int)f2bf(f.x) | ((unsigned int)f2bf(f.y) << 16);
            unsigned int p1 = (unsigned int)f2bf(f.z) | ((unsigned int)f2bf(f.w) << 16);
            *(uint2*)&Bs[row][c4 * 4] = make_uint2(p0, p1);
        }
        __syncthreads();

        Frag a[2], b[2];
#pragma unroll
        for (int mi = 0; mi < 2; ++mi) {
            const unsigned short* p = &As[wm * 32 + mi * 16 + r][0];
            a[mi].u[0] = *(const uint4*)&p[kb];
            a[mi].u[1] = *(const uint4*)&p[kb + 16];
        }
#pragma unroll
        for (int ni = 0; ni < 2; ++ni) {
            const unsigned short* p = &Bs[wn * 32 + ni * 16 + r][0];
            b[ni].u[0] = *(const uint4*)&p[kb];
            b[ni].u[1] = *(const uint4*)&p[kb + 16];
        }
#pragma unroll
        for (int mi = 0; mi < 2; ++mi)
#pragma unroll
            for (int ni = 0; ni < 2; ++ni)
                acc[mi][ni] = wmma_bf16(a[mi], b[ni], acc[mi][ni]);
    }

    const int mbase = M0 + wm * 32 + ((lane >> 4) << 3);
#pragma unroll
    for (int mi = 0; mi < 2; ++mi)
#pragma unroll
        for (int ni = 0; ni < 2; ++ni) {
            int n = N0 + wn * 32 + ni * 16 + (lane & 15);
            float bv = bias[n];
            int h = n >> 6, dloc = n & 63;
#pragma unroll
            for (int j = 0; j < 8; ++j) {
                int m = mbase + mi * 16 + j;
                int bb = m >> 9, t = m & 511;
                dst[((size_t)((bb * HH + h) * TT + t)) * 128 + hoff + dloc] =
                    f2bf(acc[mi][ni][j] + bv);
            }
        }
}

// ---------------------------------------------------------------------------
// Stage 2: flash attention per (b,h) over 128-dim heads.
// grid = 256*(T/128) = 1024, block = 256 (8 waves x 16 query rows).
// ---------------------------------------------------------------------------
__global__ __launch_bounds__(256)
void attn_kernel(const unsigned short* __restrict__ Qg,
                 const unsigned short* __restrict__ Kg,
                 const unsigned short* __restrict__ Vg,
                 unsigned short* __restrict__ Og)
{
    extern __shared__ unsigned short sm[];
    unsigned short* Qs = sm;                  // [128][136]
    unsigned short* Ks = Qs + 128 * 136;      // [64][136]
    unsigned short* Vt = Ks + 64 * 136;       // [128][72]  (transposed V tile)
    unsigned short* Ps = Vt + 128 * 72;       // [8 waves][16][72]

    const int bh = blockIdx.x >> 2;
    const int t0 = (blockIdx.x & 3) * 128;
    const int b_ = bh >> 3;
    const int h_ = bh & 7;

    const unsigned short* Qb = Qg + (size_t)bh * TT * 128;
    const unsigned short* Kb = Kg + (size_t)bh * TT * 128;
    const unsigned short* Vb = Vg + (size_t)bh * TT * 128;

    const int tid  = threadIdx.x;
    const int lane = tid & 31;
    const int wid  = tid >> 5;
    const int wq0  = wid * 16;

    const int r   = lane & 15;
    const int hi  = lane >> 4;
    const int kb  = hi * 8;
    const int kb2 = hi * 16;

#pragma unroll
    for (int i = 0; i < 8; ++i) {             // stage Q tile (128x128 bf16)
        int idx = tid + i * 256;
        int row = idx >> 4, c8 = idx & 15;
        *(uint4*)&Qs[row * 136 + c8 * 8] =
            *(const uint4*)(Qb + (size_t)(t0 + row) * 128 + c8 * 8);
    }

    const v8f vz = {0.f,0.f,0.f,0.f,0.f,0.f,0.f,0.f};
    v8f o[8];
    float mrun[8], lrun[8];
#pragma unroll
    for (int nt = 0; nt < 8; ++nt) o[nt] = vz;
#pragma unroll
    for (int j = 0; j < 8; ++j) { mrun[j] = -3.0e38f; lrun[j] = 0.f; }

    unsigned short* Pw = Ps + wid * 16 * 72;

    for (int kt = 0; kt < 8; ++kt) {
        __syncthreads();
#pragma unroll
        for (int i = 0; i < 4; ++i) {         // K tile 64x128
            int idx = tid + i * 256;
            int row = idx >> 4, c8 = idx & 15;
            *(uint4*)&Ks[row * 136 + c8 * 8] =
                *(const uint4*)(Kb + (size_t)(kt * 64 + row) * 128 + c8 * 8);
        }
#pragma unroll
        for (int i = 0; i < 4; ++i) {         // V tile, transposed into Vt[d][k]
            int idx = tid + i * 256;
            int row = idx >> 4, c8 = idx & 15;
            union { uint4 u; unsigned short s[8]; } tmp;
            tmp.u = *(const uint4*)(Vb + (size_t)(kt * 64 + row) * 128 + c8 * 8);
#pragma unroll
            for (int j2 = 0; j2 < 8; ++j2)
                Vt[(c8 * 8 + j2) * 72 + row] = tmp.s[j2];
        }
        __syncthreads();

        // S = Q K^T (16 x 64), fp32 accum
        v8f s[4] = {vz, vz, vz, vz};
#pragma unroll
        for (int ks = 0; ks < 4; ++ks) {
            Frag a;
            const unsigned short* p = &Qs[(wq0 + r) * 136 + ks * 32];
            a.u[0] = *(const uint4*)&p[kb];
            a.u[1] = *(const uint4*)&p[kb + 16];
#pragma unroll
            for (int nt = 0; nt < 4; ++nt) {
                Frag bf;
                const unsigned short* pb = &Ks[(nt * 16 + r) * 136 + ks * 32];
                bf.u[0] = *(const uint4*)&pb[kb];
                bf.u[1] = *(const uint4*)&pb[kb + 16];
                s[nt] = wmma_bf16(a, bf, s[nt]);
            }
        }

        // online softmax (row stats via half-wave shuffles)
        float mnew[8], alpha[8], rs[8];
#pragma unroll
        for (int j = 0; j < 8; ++j) {
            float mx = -3.0e38f;
#pragma unroll
            for (int nt = 0; nt < 4; ++nt) { s[nt][j] *= 0.125f; mx = fmaxf(mx, s[nt][j]); }
#pragma unroll
            for (int off = 1; off < 16; off <<= 1) mx = fmaxf(mx, __shfl_xor(mx, off, 32));
            mnew[j]  = fmaxf(mrun[j], mx);
            alpha[j] = __expf(mrun[j] - mnew[j]);
            rs[j]    = 0.f;
        }
#pragma unroll
        for (int nt = 0; nt < 4; ++nt) {
            int ccol = nt * 16 + r;
#pragma unroll
            for (int j = 0; j < 8; ++j) {
                float pv = __expf(s[nt][j] - mnew[j]);
                rs[j] += pv;
                Pw[(j + hi * 8) * 72 + ccol] = f2bf(pv);
            }
        }
#pragma unroll
        for (int j = 0; j < 8; ++j) {
            float t = rs[j];
#pragma unroll
            for (int off = 1; off < 16; off <<= 1) t += __shfl_xor(t, off, 32);
            lrun[j] = lrun[j] * alpha[j] + t;
            mrun[j] = mnew[j];
#pragma unroll
            for (int nt = 0; nt < 8; ++nt) o[nt][j] *= alpha[j];
        }
        __syncthreads();

        // O += P @ V  (contraction over 64 keys)
#pragma unroll
        for (int kstep = 0; kstep < 2; ++kstep) {
            Frag a;
            const unsigned short* p = &Pw[r * 72 + kstep * 32];
            a.u[0] = *(const uint4*)&p[kb];
            a.u[1] = *(const uint4*)&p[kb + 16];
#pragma unroll
            for (int nt = 0; nt < 8; ++nt) {
                Frag bf;
                const unsigned short* pb = &Vt[(nt * 16 + r) * 72 + kstep * 32];
                bf.u[0] = *(const uint4*)&pb[kb];
                bf.u[1] = *(const uint4*)&pb[kb + 16];
                o[nt] = wmma_bf16(a, bf, o[nt]);
            }
        }
    }

    // normalize + write bf16 into (B,T,1024) head-major layout
#pragma unroll
    for (int j = 0; j < 8; ++j) {
        float inv = 1.0f / lrun[j];
        int trow = t0 + wq0 + j + hi * 8;
        size_t base = ((size_t)(b_ * TT) + trow) * 1024 + h_ * 128;
#pragma unroll
        for (int nt = 0; nt < 8; ++nt)
            Og[base + nt * 16 + r] = f2bf(o[nt][j] * inv);
    }
}

// ---------------------------------------------------------------------------
// Stage 3: output projections. Y[g] = O[:, g*512:(g+1)*512] @ wo^T + bo.
// grid = (1024, 2), block = 256.  fp32 output straight to d_out.
// ---------------------------------------------------------------------------
__global__ __launch_bounds__(256)
void out_proj_kernel(const unsigned short* __restrict__ Om,
                     const float* __restrict__ wo1, const float* __restrict__ bo1,
                     const float* __restrict__ wo2, const float* __restrict__ bo2,
                     float* __restrict__ out)
{
    __shared__ __align__(16) unsigned short As[128][40];
    __shared__ __align__(16) unsigned short Bs[64][40];

    const int g = blockIdx.y;
    const float* W    = g ? wo2 : wo1;
    const float* bias = g ? bo2 : bo1;
    float* dst = out + (size_t)g * BT * DD;
    const int colbase = g * DD;

    const int M0 = (blockIdx.x >> 3) * 128;
    const int N0 = (blockIdx.x & 7) * 64;

    const int tid  = threadIdx.x;
    const int lane = tid & 31;
    const int wid  = tid >> 5;
    const int wm   = wid >> 1;
    const int wn   = wid & 1;

    const v8f vz = {0.f,0.f,0.f,0.f,0.f,0.f,0.f,0.f};
    v8f acc[2][2] = {{vz, vz}, {vz, vz}};

    const int r   = lane & 15;
    const int kb  = (lane >> 4) * 8;
    const int kb2 = (lane >> 4) * 16;

    for (int k0 = 0; k0 < DD; k0 += 32) {
        __syncthreads();
#pragma unroll
        for (int i = 0; i < 2; ++i) {         // A: already bf16 in workspace
            int idx = tid + i * 256;
            int row = idx >> 2, c8 = idx & 3;
            *(uint4*)&As[row][c8 * 8] =
                *(const uint4*)(Om + (size_t)(M0 + row) * 1024 + colbase + k0 + c8 * 8);
        }
#pragma unroll
        for (int i = 0; i < 2; ++i) {         // B: wo fp32 -> bf16
            int idx = tid + i * 256;
            int row = idx >> 3, c4 = idx & 7;
            float4 f = *(const float4*)(W + (size_t)(N0 + row) * DD + k0 + c4 * 4);
            unsigned int p0 = (unsigned int)f2bf(f.x) | ((unsigned int)f2bf(f.y) << 16);
            unsigned int p1 = (unsigned int)f2bf(f.z) | ((unsigned int)f2bf(f.w) << 16);
            *(uint2*)&Bs[row][c4 * 4] = make_uint2(p0, p1);
        }
        __syncthreads();

        Frag a[2], b[2];
#pragma unroll
        for (int mi = 0; mi < 2; ++mi) {
            const unsigned short* p = &As[wm * 32 + mi * 16 + r][0];
            a[mi].u[0] = *(const uint4*)&p[kb];
            a[mi].u[1] = *(const uint4*)&p[kb + 16];
        }
#pragma unroll
        for (int ni = 0; ni < 2; ++ni) {
            const unsigned short* p = &Bs[wn * 32 + ni * 16 + r][0];
            b[ni].u[0] = *(const uint4*)&p[kb];
            b[ni].u[1] = *(const uint4*)&p[kb + 16];
        }
#pragma unroll
        for (int mi = 0; mi < 2; ++mi)
#pragma unroll
            for (int ni = 0; ni < 2; ++ni)
                acc[mi][ni] = wmma_bf16(a[mi], b[ni], acc[mi][ni]);
    }

    const int mbase = M0 + wm * 32 + ((lane >> 4) << 3);
#pragma unroll
    for (int mi = 0; mi < 2; ++mi)
#pragma unroll
        for (int ni = 0; ni < 2; ++ni) {
            int n = N0 + wn * 32 + ni * 16 + (lane & 15);
            float bv = bias[n];
#pragma unroll
            for (int j = 0; j < 8; ++j) {
                int m = mbase + mi * 16 + j;
                dst[(size_t)m * DD + n] = acc[mi][ni][j] + bv;
            }
        }
}

// ---------------------------------------------------------------------------
extern "C" void kernel_launch(void* const* d_in, const int* in_sizes, int n_in,
                              void* d_out, int out_size, void* d_ws, size_t ws_size,
                              hipStream_t stream)
{
    (void)in_sizes; (void)n_in; (void)out_size; (void)ws_size;

    const float* xf  = (const float*)d_in[0];
    const float* xs  = (const float*)d_in[1];
    const float* wq1 = (const float*)d_in[2];  const float* bq1 = (const float*)d_in[3];
    const float* wk1 = (const float*)d_in[4];  const float* bk1 = (const float*)d_in[5];
    const float* wv1 = (const float*)d_in[6];  const float* bv1 = (const float*)d_in[7];
    const float* wq2 = (const float*)d_in[8];  const float* bq2 = (const float*)d_in[9];
    const float* wk2 = (const float*)d_in[10]; const float* bk2 = (const float*)d_in[11];
    const float* wv2 = (const float*)d_in[12]; const float* bv2 = (const float*)d_in[13];
    const float* wo1 = (const float*)d_in[14]; const float* bo1 = (const float*)d_in[15];
    const float* wo2 = (const float*)d_in[16]; const float* bo2 = (const float*)d_in[17];

    unsigned short* ws = (unsigned short*)d_ws;
    const size_t QKV = (size_t)BB * HH * TT * 128;   // 16,777,216 elems (32 MB bf16)
    unsigned short* q = ws;
    unsigned short* k = ws + QKV;
    unsigned short* v = ws + 2 * QKV;
    unsigned short* O = ws + 3 * QKV;

    qkv_proj_kernel<<<dim3(1024, 6), 256, 0, stream>>>(
        xf, xs, wq1, wk1, wv1, wq2, wk2, wv2,
        bq1, bk1, bv1, bq2, bk2, bv2, q, k, v);

    size_t smem = (size_t)(128 * 136 + 64 * 136 + 128 * 72 + 8 * 16 * 72)
                  * sizeof(unsigned short);
    attn_kernel<<<dim3(1024), 256, smem, stream>>>(q, k, v, O);

    out_proj_kernel<<<dim3(1024, 2), 256, 0, stream>>>(O, wo1, bo1, wo2, bo2,
                                                       (float*)d_out);
}
